// BahdanauAttention_64742337020204
// MI455X (gfx1250) — matrix operand order repaired and checked by reading, not compiled, not run on hardware
//
#include <hip/hip_runtime.h>

#define D2 1024
#define SEQ 2048
#define NBATCH 32
#define MTILE 32
#define LDSTRIDE 1032  // halfs per LDS row (+8 pad -> 4-bank rotation per row)

typedef __attribute__((ext_vector_type(16))) __bf16 v16bf;
typedef __attribute__((ext_vector_type(8)))  float  v8f;

union ABFrag { uint4 q[2]; v16bf v; };

__device__ __forceinline__ unsigned short f2bf(float f) {
  unsigned int u = __float_as_uint(f);
  u += 0x7FFFu + ((u >> 16) & 1u);   // round-to-nearest-even
  return (unsigned short)(u >> 16);
}

// ---- 1. Ua_w fp32 -> bf16 (2 MB, stays resident in L2) ----
__global__ __launch_bounds__(256)
void cvt_ua_kernel(const float* __restrict__ ua, unsigned short* __restrict__ out) {
  int i = blockIdx.x * 256 + threadIdx.x;          // one float4 per thread
  float4 v = ((const float4*)ua)[i];
  ushort4 o;
  o.x = f2bf(v.x); o.y = f2bf(v.y); o.z = f2bf(v.z); o.w = f2bf(v.w);
  ((ushort4*)out)[i] = o;
}

// ---- 2. qq[b][j] = Wa_b[j] + Ua_b[j] + q_cat[b] . Wa_w[j,:]  ----
__global__ __launch_bounds__(256)
void qw_kernel(const float* __restrict__ query, const float* __restrict__ Wa_w,
               const float* __restrict__ Wa_b, const float* __restrict__ Ua_b,
               float* __restrict__ qq) {
  __shared__ float qs[D2];
  const int b = blockIdx.x >> 2;
  const int j = (blockIdx.x & 3) * 256 + threadIdx.x;
  // q_cat[b] = [query[b,1,:], query[b,3,:]]  (query is (B,4,512))
  const float* qbase = query + (size_t)b * 2048 + 512;
  for (int k = threadIdx.x; k < D2; k += 256)
    qs[k] = qbase[(k < 512) ? k : (k + 512)];
  __syncthreads();
  const float4* wr = (const float4*)(Wa_w + (size_t)j * D2);
  float acc = 0.f;
#pragma unroll 4
  for (int k4 = 0; k4 < D2 / 4; ++k4) {
    float4 w = wr[k4];
    const float* q4 = &qs[k4 * 4];
    acc += w.x * q4[0] + w.y * q4[1] + w.z * q4[2] + w.w * q4[3];
  }
  qq[(size_t)b * D2 + j] = acc + Wa_b[j] + Ua_b[j];
}

// ---- 3. fused  scores[b,s] = Va_b + sum_j Va_w[j]*tanh(qq[b,j] + keys[b,s,:].Ua_w[j,:])
// One workgroup: 32 key rows x full 1024 j-dim; 8 waves each own 8 N-tiles of 16.
// K is the OUTER compute loop: A fragments loaded once per k-step and reused by
// 4 N-tiles (x2 M-subtiles), keeping register pressure bounded (no spills).
__global__ __launch_bounds__(256)
void scores_kernel(const float* __restrict__ keys, const unsigned short* __restrict__ uab,
                   const float* __restrict__ qq, const float* __restrict__ Va_w,
                   const float* __restrict__ Va_b, float* __restrict__ scores) {
  __shared__ unsigned short tileA[MTILE * LDSTRIDE];   // 66 KB bf16 keys tile
  __shared__ float swred[8][MTILE];

  const int tid = threadIdx.x;
  const int b   = blockIdx.x >> 6;           // SEQ/MTILE = 64 tiles per batch
  const int s0  = (blockIdx.x & 63) * MTILE;

  // stage keys tile fp32 -> bf16 into LDS (coalesced float4 reads)
  const float4* src = (const float4*)(keys + ((size_t)b * SEQ + s0) * D2);
  for (int i = tid; i < MTILE * D2 / 4; i += 256) {
    float4 v = src[i];
    int r = i >> 8;                // D2/4 = 256 float4 per row
    int c = i & 255;
    uint2 pk;
    pk.x = (unsigned)f2bf(v.x) | ((unsigned)f2bf(v.y) << 16);
    pk.y = (unsigned)f2bf(v.z) | ((unsigned)f2bf(v.w) << 16);
    *(uint2*)&tileA[r * LDSTRIDE + c * 4] = pk;
  }
  __syncthreads();

  const int wave  = tid >> 5;
  const int lane  = tid & 31;
  const int nlo   = lane & 15;
  const int hi    = lane >> 4;
  const int koffA = hi * 8;        // A frag: K {0..7,16..23} vs {8..15,24..31}
  const int koffB = hi * 8;        // B frag: K {0..7,16..23} vs {8..15,24..31}

  float rp0[8], rp1[8];
#pragma unroll
  for (int i = 0; i < 8; ++i) { rp0[i] = 0.f; rp1[i] = 0.f; }

  const unsigned short* arow0 = &tileA[nlo * LDSTRIDE + koffA];
  const unsigned short* arow1 = &tileA[(nlo + 16) * LDSTRIDE + koffA];

#pragma unroll 1
  for (int g = 0; g < 2; ++g) {
    // base row of Ua (bf16) for this lane's column in each of the 4 N-tiles
    const unsigned short* brow =
        uab + (size_t)((wave * 8 + g * 4) * 16 + nlo) * D2 + koffB;

    v8f acc0[4], acc1[4];
#pragma unroll
    for (int nt = 0; nt < 4; ++nt) { acc0[nt] = (v8f){}; acc1[nt] = (v8f){}; }

#pragma unroll 2
    for (int k0 = 0; k0 < D2; k0 += 32) {
      ABFrag a0, a1;
      const uint4* ap0 = (const uint4*)(arow0 + k0);
      a0.q[0] = ap0[0];            // K koffA .. +7
      a0.q[1] = ap0[2];            // K koffA+16 .. +23
      const uint4* ap1 = (const uint4*)(arow1 + k0);
      a1.q[0] = ap1[0];
      a1.q[1] = ap1[2];
#pragma unroll
      for (int nt = 0; nt < 4; ++nt) {
        ABFrag bfr;
        const uint4* bp = (const uint4*)(brow + (size_t)nt * 16 * D2 + k0);
        bfr.q[0] = bp[0];          // K koffB .. +7, then K koffB+16 .. +23
        bfr.q[1] = bp[2];
        acc0[nt] = __builtin_amdgcn_wmma_f32_16x16x32_bf16(false, a0.v, false, bfr.v,
                                                           (short)0, acc0[nt], false, false);
        acc1[nt] = __builtin_amdgcn_wmma_f32_16x16x32_bf16(false, a1.v, false, bfr.v,
                                                           (short)0, acc1[nt], false, false);
      }
    }

    // epilogue: C layout n = lane&15, m = i + 8*hi
#pragma unroll
    for (int nt = 0; nt < 4; ++nt) {
      const int j = (wave * 8 + g * 4 + nt) * 16 + nlo;
      const float qv = qq[(size_t)b * D2 + j];
      const float vw = Va_w[j];
#pragma unroll
      for (int i = 0; i < 8; ++i) {
        rp0[i] += vw * tanhf(acc0[nt][i] + qv);
        rp1[i] += vw * tanhf(acc1[nt][i] + qv);
      }
    }
  }

  // reduce over the 16-lane N dimension (stays within each 16-lane half)
#pragma unroll
  for (int m = 1; m < 16; m <<= 1) {
#pragma unroll
    for (int i = 0; i < 8; ++i) {
      rp0[i] += __shfl_xor(rp0[i], m, 32);
      rp1[i] += __shfl_xor(rp1[i], m, 32);
    }
  }
  if (nlo == 0) {
    const int mbase = hi * 8;
#pragma unroll
    for (int i = 0; i < 8; ++i) {
      swred[wave][mbase + i]      = rp0[i];
      swred[wave][16 + mbase + i] = rp1[i];
    }
  }
  __syncthreads();
  if (wave == 0) {                 // 32 lanes -> 32 rows
    float ssum = Va_b[0];
#pragma unroll
    for (int w = 0; w < 8; ++w) ssum += swred[w][lane];
    scores[(size_t)b * SEQ + s0 + lane] = ssum;
  }
}

// ---- 4. softmax over S, in place in the weights region ----
__global__ __launch_bounds__(256)
void softmax_kernel(float* __restrict__ w) {
  __shared__ float red[256];
  const int b = blockIdx.x, tid = threadIdx.x;
  float* row = w + (size_t)b * SEQ;
  float v[8];
  float mx = -3.4e38f;
#pragma unroll
  for (int i = 0; i < 8; ++i) { v[i] = row[tid + i * 256]; mx = fmaxf(mx, v[i]); }
  red[tid] = mx; __syncthreads();
  for (int s = 128; s > 0; s >>= 1) {
    if (tid < s) red[tid] = fmaxf(red[tid], red[tid + s]);
    __syncthreads();
  }
  mx = red[0]; __syncthreads();
  float sum = 0.f;
#pragma unroll
  for (int i = 0; i < 8; ++i) { v[i] = __expf(v[i] - mx); sum += v[i]; }
  red[tid] = sum; __syncthreads();
  for (int s = 128; s > 0; s >>= 1) {
    if (tid < s) red[tid] += red[tid + s];
    __syncthreads();
  }
  const float inv = 1.f / red[0];
#pragma unroll
  for (int i = 0; i < 8; ++i) row[tid + i * 256] = v[i] * inv;
}

// ---- 5. context[b,d] = sum_s weights[b,s]*keys[b,s,d]  (HBM bandwidth bound) ----
__global__ __launch_bounds__(256)
void context_kernel(const float* __restrict__ keys, const float* __restrict__ w,
                    float* __restrict__ ctx) {
  const int b = blockIdx.y;
  const int d = blockIdx.x * 256 + threadIdx.x;
  const float* kb = keys + (size_t)b * SEQ * D2 + d;
  const float* wb = w + (size_t)b * SEQ;
  float acc = 0.f;
#pragma unroll 4
  for (int s = 0; s < SEQ; ++s) acc += wb[s] * kb[(size_t)s * D2];
  ctx[(size_t)b * D2 + d] = acc;
}

extern "C" void kernel_launch(void* const* d_in, const int* in_sizes, int n_in,
                              void* d_out, int out_size, void* d_ws, size_t ws_size,
                              hipStream_t stream) {
  const float* query = (const float*)d_in[0];
  const float* keys  = (const float*)d_in[1];
  const float* Wa_w  = (const float*)d_in[2];
  const float* Wa_b  = (const float*)d_in[3];
  const float* Ua_w  = (const float*)d_in[4];
  const float* Ua_b  = (const float*)d_in[5];
  const float* Va_w  = (const float*)d_in[6];
  const float* Va_b  = (const float*)d_in[7];

  float* ctx = (float*)d_out;                      // (32,1,1024)
  float* wts = (float*)d_out + NBATCH * D2;        // (32,1,2048): scores -> weights in place

  unsigned short* uab = (unsigned short*)d_ws;                       // 2 MB bf16 Ua_w
  float* qq = (float*)((char*)d_ws + (size_t)D2 * D2 * 2);           // 128 KB

  cvt_ua_kernel<<<(D2 * D2) / 1024, 256, 0, stream>>>(Ua_w, uab);
  qw_kernel<<<NBATCH * (D2 / 256), 256, 0, stream>>>(query, Wa_w, Wa_b, Ua_b, qq);
  scores_kernel<<<NBATCH * (SEQ / MTILE), 256, 0, stream>>>(keys, uab, qq, Va_w, Va_b, wts);
  softmax_kernel<<<NBATCH, 256, 0, stream>>>(wts);
  context_kernel<<<dim3(D2 / 256, NBATCH), 256, 0, stream>>>(keys, wts, ctx);
}
